// GraphAttentionLayer2D_54262616818212
// MI455X (gfx1250) — compile-verified, matrix operand order checked by reading, not run on hardware
//
#include <hip/hip_runtime.h>
#include <hip/hip_bf16.h>
#include <math.h>

// ---------------------------------------------------------------------------
// Problem constants: h:(64,256,256,7) W:(256,256) a:(512,1) B:(7,7)
// GEMM: M = 64*7*256 = 114688, K = T = 256, N = F = 256.
// ---------------------------------------------------------------------------
#define N_B   64
#define C_DIM 256
#define T_DIM 256
#define V_N   7
#define F_DIM 256
#define M_DIM (N_B * V_N * C_DIM)   // 114688

typedef __attribute__((ext_vector_type(16))) __bf16 v16bf;
typedef __attribute__((ext_vector_type(8)))  __bf16 v8bf;
typedef __attribute__((ext_vector_type(8)))  float  v8f;

// ---------------------------------------------------------------------------
// K0: repack W into fragment-major bf16 (ISA 16-bit B layout, per-lane
// contiguous 16 elements), and compute adj_norm (7x7) once.
// Wfrag[((ks*16 + ft)*32 + lane)*16 + j] = W[ks*32 + kOff(lane) + kj(j),
//                                           ft*16 + (lane&15)]
// ---------------------------------------------------------------------------
__global__ __launch_bounds__(256) void k_prep(const float* __restrict__ W,
                                              const float* __restrict__ B,
                                              __bf16* __restrict__ Wfrag,
                                              float* __restrict__ adjn) {
  const int t = blockIdx.x * 256 + threadIdx.x;   // 0..4095
  if (t < 8 * 16 * 32) {
    const int lane = t & 31;
    const int ft   = (t >> 5) & 15;
    const int ks   = t >> 9;
    const int kOff = (lane >> 4) * 8;
    const int col  = ft * 16 + (lane & 15);
    __bf16* dst = Wfrag + (size_t)t * 16;
#pragma unroll
    for (int j = 0; j < 16; ++j) {
      const int k = ks * 32 + kOff + ((j < 8) ? j : (j + 8));
      dst[j] = (__bf16)W[k * F_DIM + col];
    }
  }

  if (blockIdx.x == 0 && threadIdx.x == 0) {
    float adj[49];
    float mn = 1e30f, mx = -1e30f;
    for (int i = 0; i < 7; ++i)
      for (int j = 0; j < 7; ++j) {
        float x = B[i * 7 + j] + ((i == j) ? 1.0f : 0.0f);
        adj[i * 7 + j] = x;
        mn = fminf(mn, x);
        mx = fmaxf(mx, x);
      }
    const float inv = 1.0f / (mx - mn);
    float dinv[7];
    for (int i = 0; i < 7; ++i) {
      float s = 0.f;
      for (int j = 0; j < 7; ++j) {
        adj[i * 7 + j] = (adj[i * 7 + j] - mn) * inv;
        s += adj[i * 7 + j];
      }
      dinv[i] = 1.0f / sqrtf(s);
    }
    for (int i = 0; i < 7; ++i)
      for (int j = 0; j < 7; ++j)
        adjn[i * 7 + j] = adj[i * 7 + j] * dinv[i] * dinv[j];
  }
}

// ---------------------------------------------------------------------------
// K1: transpose-pack h (N,C,T,V) f32 -> Abf[m, t] bf16 row-major
// (m = (n*7+v)*256 + c).  Coalesced reads, LDS transpose (stride 7 ->
// conflict-free mod 64 banks), coalesced 512B writes per (n,v,c) row.
// ---------------------------------------------------------------------------
__global__ __launch_bounds__(256) void k_packA(const float* __restrict__ h,
                                               __bf16* __restrict__ Abf) {
  const int b = blockIdx.x;           // n*256 + c
  const int n = b >> 8, c = b & 255;
  __shared__ float tile[T_DIM * V_N]; // 1792 floats
  const float* src = h + (size_t)b * (T_DIM * V_N);
  for (int i = threadIdx.x; i < T_DIM * V_N; i += 256) tile[i] = src[i];
  __syncthreads();
  const int t = threadIdx.x;          // 0..255
#pragma unroll
  for (int v = 0; v < V_N; ++v) {
    const size_t m = (size_t)(n * V_N + v) * C_DIM + c;
    Abf[m * T_DIM + t] = (__bf16)tile[t * V_N + v];
  }
}

// ---------------------------------------------------------------------------
// K2: Wh = Abf @ W  via v_wmma_f32_16x16x32_bf16.
// 256 threads = 8 wave32s as 2(M) x 4(F); wave tile 32 x 64 -> 8 accumulators.
// Block tile 64(M) x 256(F); grid = M/64 = 1792.
// A fragments: two contiguous 16B loads, concat via shufflevector.
// B fragments: one contiguous 32B load from fragment-major Wfrag.
// ---------------------------------------------------------------------------
#define WMMA_BF16(A, Bm, Cm) \
  __builtin_amdgcn_wmma_f32_16x16x32_bf16(false, A, false, Bm, (short)0, Cm, false, false)

__global__ __launch_bounds__(256) void k_gemm_wh(const __bf16* __restrict__ Abf,
                                                 const __bf16* __restrict__ Wfrag,
                                                 float* __restrict__ Wh) {
  const int lane   = threadIdx.x & 31;
  const int wave   = threadIdx.x >> 5;
  const int mWave  = blockIdx.x * 64 + (wave & 1) * 32;
  const int fTile0 = (wave >> 1) * 4;         // first of 4 16-col tiles
  const int row16  = lane & 15;
  const int kOff   = (lane >> 4) * 8;

  const __bf16* aRow0 = Abf + (size_t)(mWave + row16) * T_DIM + kOff;
  const __bf16* aRow1 = aRow0 + (size_t)16 * T_DIM;

  v8f acc0 = {}, acc1 = {}, acc2 = {}, acc3 = {};
  v8f acc4 = {}, acc5 = {}, acc6 = {}, acc7 = {};

#pragma unroll 2
  for (int ks = 0; ks < 8; ++ks) {
    const int kk = ks * 32;
    if (ks < 7) {
      __builtin_prefetch(aRow0 + kk + 32, 0, 1);
      __builtin_prefetch(aRow1 + kk + 32, 0, 1);
    }
    const v8bf lo0 = *(const v8bf*)(aRow0 + kk);
    const v8bf hi0 = *(const v8bf*)(aRow0 + kk + 16);
    const v8bf lo1 = *(const v8bf*)(aRow1 + kk);
    const v8bf hi1 = *(const v8bf*)(aRow1 + kk + 16);
    const v16bf a0 = __builtin_shufflevector(lo0, hi0, 0, 1, 2, 3, 4, 5, 6, 7,
                                             8, 9, 10, 11, 12, 13, 14, 15);
    const v16bf a1 = __builtin_shufflevector(lo1, hi1, 0, 1, 2, 3, 4, 5, 6, 7,
                                             8, 9, 10, 11, 12, 13, 14, 15);

    const __bf16* wbase =
        Wfrag + (((size_t)ks * 16 + fTile0) * 32 + lane) * 16;
    const v16bf b0 = *(const v16bf*)(wbase);
    const v16bf b1 = *(const v16bf*)(wbase + 32 * 16);
    const v16bf b2 = *(const v16bf*)(wbase + 64 * 16);
    const v16bf b3 = *(const v16bf*)(wbase + 96 * 16);

    acc0 = WMMA_BF16(a0, b0, acc0);
    acc1 = WMMA_BF16(a0, b1, acc1);
    acc2 = WMMA_BF16(a0, b2, acc2);
    acc3 = WMMA_BF16(a0, b3, acc3);
    acc4 = WMMA_BF16(a1, b0, acc4);
    acc5 = WMMA_BF16(a1, b1, acc5);
    acc6 = WMMA_BF16(a1, b2, acc6);
    acc7 = WMMA_BF16(a1, b3, acc7);
  }

  // C/D layout: VGPR r -> M = r (+8 for lanes 16-31), N = lane&15.
  const int rowHi = (lane >> 4) * 8;
  const int col   = fTile0 * 16 + (lane & 15);
#pragma unroll
  for (int r = 0; r < 8; ++r) {
    float* o0 = Wh + (size_t)(mWave + rowHi + r) * F_DIM + col;
    o0[0]  = acc0[r];
    o0[16] = acc1[r];
    o0[32] = acc2[r];
    o0[48] = acc3[r];
    float* o1 = o0 + (size_t)16 * F_DIM;
    o1[0]  = acc4[r];
    o1[16] = acc5[r];
    o1[32] = acc6[r];
    o1[48] = acc7[r];
  }
}

// ---------------------------------------------------------------------------
// K3: s1[m] = Wh[m,:] . a1 ; s2[m] = Wh[m,:] . a2   (one wave32 per row)
// ---------------------------------------------------------------------------
__global__ __launch_bounds__(256) void k_s12(const float* __restrict__ Wh,
                                             const float* __restrict__ a,
                                             float* __restrict__ s1,
                                             float* __restrict__ s2) {
  const int lane = threadIdx.x & 31;
  const int wave = threadIdx.x >> 5;
  const size_t row = (size_t)blockIdx.x * 8 + wave;
  const float* r = Wh + row * F_DIM;
  float sum1 = 0.f, sum2 = 0.f;
#pragma unroll
  for (int j = 0; j < 8; ++j) {
    const int f = lane + 32 * j;
    const float w = r[f];
    sum1 += w * a[f];
    sum2 += w * a[F_DIM + f];
  }
#pragma unroll
  for (int off = 16; off > 0; off >>= 1) {
    sum1 += __shfl_down(sum1, off, 32);
    sum2 += __shfl_down(sum2, off, 32);
  }
  if (lane == 0) { s1[row] = sum1; s2[row] = sum2; }
}

// ---------------------------------------------------------------------------
// K4: softmax stats over the C axis for each (n,i,j).
// ---------------------------------------------------------------------------
__global__ __launch_bounds__(256) void k_softstat(const float* __restrict__ s1,
                                                  const float* __restrict__ s2,
                                                  float* __restrict__ emax,
                                                  float* __restrict__ esum) {
  const int b  = blockIdx.x;          // n*49 + i*7 + j
  const int n  = b / 49;
  const int ij = b % 49;
  const int i  = ij / 7, j = ij % 7;
  const int c  = threadIdx.x;

  float e = s1[(size_t)(n * 7 + i) * C_DIM + c] + s2[(size_t)(n * 7 + j) * C_DIM + c];
  e = (e > 0.f) ? e : 0.2f * e;

  __shared__ float red[256];
  red[c] = e;
  __syncthreads();
  for (int s = 128; s > 0; s >>= 1) {
    if (c < s) red[c] = fmaxf(red[c], red[c + s]);
    __syncthreads();
  }
  const float mx = red[0];
  __syncthreads();
  red[c] = expf(e - mx);
  __syncthreads();
  for (int s = 128; s > 0; s >>= 1) {
    if (c < s) red[c] += red[c + s];
    __syncthreads();
  }
  if (c == 0) { emax[b] = mx; esum[b] = red[0]; }
}

// ---------------------------------------------------------------------------
// K5: one block per (n,c):
//   att[i,j] = exp(leaky(s1+s2) - emax)/esum
//   w2[j,v]  = sum_i adjn[i,v] * att[i,j]
//   out[n,c,f,v] = elu( sum_j Wh[n,j,c,f] * w2[j,v] )
// ---------------------------------------------------------------------------
__global__ __launch_bounds__(256) void k_out(const float* __restrict__ Wh,
                                             const float* __restrict__ s1,
                                             const float* __restrict__ s2,
                                             const float* __restrict__ emax,
                                             const float* __restrict__ esum,
                                             const float* __restrict__ adjn,
                                             float* __restrict__ out) {
  const int b = blockIdx.x;           // n*256 + c
  const int n = b >> 8;
  const int c = b & 255;
  const int tid = threadIdx.x;

  __shared__ float att_s[49];
  __shared__ float adj_s[49];
  __shared__ float w2_s[49];          // [j*7 + v]

  if (tid < 49) {
    const int i = tid / 7, j = tid % 7;
    float e = s1[(size_t)(n * 7 + i) * C_DIM + c] + s2[(size_t)(n * 7 + j) * C_DIM + c];
    e = (e > 0.f) ? e : 0.2f * e;
    const int nij = n * 49 + tid;
    att_s[tid] = expf(e - emax[nij]) / esum[nij];
    adj_s[tid] = adjn[tid];
  }
  __syncthreads();
  if (tid < 49) {
    const int j = tid / 7, v = tid % 7;
    float s = 0.f;
#pragma unroll
    for (int i = 0; i < 7; ++i) s += adj_s[i * 7 + v] * att_s[i * 7 + j];
    w2_s[j * 7 + v] = s;
  }
  __syncthreads();

  const int f = tid;                  // 0..255
  float wh[V_N];
#pragma unroll
  for (int j = 0; j < V_N; ++j)
    wh[j] = Wh[((size_t)(n * 7 + j) * C_DIM + c) * F_DIM + f];

  float* o = out + ((size_t)b * F_DIM + f) * V_N;
#pragma unroll
  for (int v = 0; v < V_N; ++v) {
    float s = 0.f;
#pragma unroll
    for (int j = 0; j < V_N; ++j) s += wh[j] * w2_s[j * 7 + v];
    o[v] = (s > 0.f) ? s : expm1f(s);    // jax.nn.elu, alpha=1
  }
}

// ---------------------------------------------------------------------------
// Workspace layout (256-byte aligned regions)
// ---------------------------------------------------------------------------
#define OFF_ADJ   ((size_t)0)                    // 49 f32
#define OFF_WFRAG ((size_t)256)                  // 65536 bf16 = 131072 B
#define OFF_S1    (OFF_WFRAG + 131072)           // 114688 f32
#define OFF_S2    (OFF_S1 + 458752)              // 114688 f32
#define OFF_EMAX  (OFF_S2 + 458752)              // 3136 f32
#define OFF_ESUM  (OFF_EMAX + 12544)             // 3136 f32
#define OFF_ABF   (OFF_ESUM + 12544)             // 114688*256 bf16 = 58720256 B
#define OFF_WH    (OFF_ABF + 58720256)           // 114688*256 f32

extern "C" void kernel_launch(void* const* d_in, const int* in_sizes, int n_in,
                              void* d_out, int out_size, void* d_ws, size_t ws_size,
                              hipStream_t stream) {
  const float* h = (const float*)d_in[0];   // (64,256,256,7)
  const float* W = (const float*)d_in[1];   // (256,256)
  const float* a = (const float*)d_in[2];   // (512,1)
  const float* B = (const float*)d_in[3];   // (7,7)
  float* out = (float*)d_out;               // (64,256,256,7)

  char* ws = (char*)d_ws;
  float*  adjn  = (float*)(ws + OFF_ADJ);
  __bf16* Wfrag = (__bf16*)(ws + OFF_WFRAG);
  float*  s1    = (float*)(ws + OFF_S1);
  float*  s2    = (float*)(ws + OFF_S2);
  float*  emax  = (float*)(ws + OFF_EMAX);
  float*  esum  = (float*)(ws + OFF_ESUM);
  __bf16* Abf   = (__bf16*)(ws + OFF_ABF);
  float*  Wh    = (float*)(ws + OFF_WH);

  k_prep    <<<16,             256, 0, stream>>>(W, B, Wfrag, adjn);
  k_packA   <<<N_B * C_DIM,    256, 0, stream>>>(h, Abf);
  k_gemm_wh <<<M_DIM / 64,     256, 0, stream>>>(Abf, Wfrag, Wh);
  k_s12     <<<M_DIM / 8,      256, 0, stream>>>(Wh, a, s1, s2);
  k_softstat<<<N_B * 49,       256, 0, stream>>>(s1, s2, emax, esum);
  k_out     <<<N_B * C_DIM,    256, 0, stream>>>(Wh, s1, s2, emax, esum, adjn, out);
}
